// SparseAttention_31233002176611
// MI455X (gfx1250) — matrix operand order repaired and checked by reading, not compiled, not run on hardware
//
#include <hip/hip_runtime.h>
#include <math.h>

typedef _Float16 v16h __attribute__((ext_vector_type(16)));
typedef _Float16 v8h  __attribute__((ext_vector_type(8)));
typedef _Float16 v4h  __attribute__((ext_vector_type(4)));
typedef float    v8f  __attribute__((ext_vector_type(8)));
typedef unsigned int  uint4v __attribute__((ext_vector_type(4)));
typedef int           int8v  __attribute__((ext_vector_type(8)));
typedef int           int4v  __attribute__((ext_vector_type(4)));

namespace {
constexpr int kB = 2, kT = 2048, kH = 16, kN = 512;
constexpr int QT = 16;      // query rows per workgroup
constexpr int KTILE = 32;   // keys per iteration
constexpr int NWAVE = 4;    // waves per workgroup
constexpr int DW = 128;     // head dims owned per wave
constexpr float kScale      = 0.044194173824159216f;  // 1/sqrt(512)
constexpr float kNegLnTheta = -9.210340371976184f;    // -ln(10000)
constexpr float kInv2Pi     = 0.15915494309189535f;
constexpr float kTwoPi      = 6.283185307179586f;
}

__device__ __forceinline__ v8f zero8() {
    v8f z;
#pragma unroll
    for (int i = 0; i < 8; ++i) z[i] = 0.0f;
    return z;
}

__device__ __forceinline__ float pair_freq(int d) {   // d even
    return __expf((float)d * (kNegLnTheta / (float)kN)) * kInv2Pi;
}

__device__ __forceinline__ void rope_pair(float ve, float vo, float t, float f,
                                          float& oe, float& oo) {
    float ph  = t * f;
    float ang = (ph - floorf(ph)) * kTwoPi;
    float s = __sinf(ang);
    float c = __cosf(ang);
    oe = ve * c - vo * s;
    oo = vo * c + ve * s;
}

// ============================ Pass 1a: RoPE(Q,K) -> f16 [b,h,t,d] ============
__global__ __launch_bounds__(256) void rope_qk_f16(const float* __restrict__ Q,
                                                   const float* __restrict__ K,
                                                   _Float16* __restrict__ Qh,
                                                   _Float16* __restrict__ Kh) {
    const int bt = blockIdx.x;           // (b,t) pair
    const int b = bt / kT, t = bt % kT;
    const float tf = (float)t;
    const float* qrow = Q + (size_t)bt * (kH * kN);
    const float* krow = K + (size_t)bt * (kH * kN);
#pragma unroll 2
    for (int base = threadIdx.x * 4; base < kH * kN; base += 256 * 4) {
        const int h = base >> 9;
        const int d = base & (kN - 1);
        const float f0 = pair_freq(d), f1 = pair_freq(d + 2);
        float4 fq = *reinterpret_cast<const float4*>(qrow + base);
        float4 fk = *reinterpret_cast<const float4*>(krow + base);
        float a, bb, c, dd;
        v4h oq, ok;
        rope_pair(fq.x, fq.y, tf, f0, a, bb);  oq[0] = (_Float16)a; oq[1] = (_Float16)bb;
        rope_pair(fq.z, fq.w, tf, f1, c, dd);  oq[2] = (_Float16)c; oq[3] = (_Float16)dd;
        rope_pair(fk.x, fk.y, tf, f0, a, bb);  ok[0] = (_Float16)a; ok[1] = (_Float16)bb;
        rope_pair(fk.z, fk.w, tf, f1, c, dd);  ok[2] = (_Float16)c; ok[3] = (_Float16)dd;
        const size_t dst = (((size_t)b * kH + h) * kT + t) * kN + d;
        *reinterpret_cast<v4h*>(Qh + dst) = oq;
        *reinterpret_cast<v4h*>(Kh + dst) = ok;
    }
}

// ============================ Pass 1b: V -> f16 transposed [b,h,d,t] =========
__global__ __launch_bounds__(128) void v_tr_f16(const float* __restrict__ V,
                                                _Float16* __restrict__ Vth) {
    __shared__ _Float16 tile[32][33];
    const int tiles_d = kN / 32, tiles_t = kT / 32;
    int idx = blockIdx.x;
    const int td = idx % tiles_d; idx /= tiles_d;
    const int tt = idx % tiles_t; idx /= tiles_t;
    const int bh = idx;                       // b*kH + h
    const int b = bh / kH, h = bh % kH;
    const int t0 = tt * 32, d0 = td * 32;
    const int r  = threadIdx.x >> 2;          // 0..31
    const int c0 = (threadIdx.x & 3) * 8;     // 0,8,16,24
    {
        const float* src = V + ((size_t)b * kT + t0 + r) * (kH * kN) + h * kN + d0 + c0;
        float4 f0 = *reinterpret_cast<const float4*>(src);
        float4 f1 = *reinterpret_cast<const float4*>(src + 4);
        tile[r][c0 + 0] = (_Float16)f0.x; tile[r][c0 + 1] = (_Float16)f0.y;
        tile[r][c0 + 2] = (_Float16)f0.z; tile[r][c0 + 3] = (_Float16)f0.w;
        tile[r][c0 + 4] = (_Float16)f1.x; tile[r][c0 + 5] = (_Float16)f1.y;
        tile[r][c0 + 6] = (_Float16)f1.z; tile[r][c0 + 7] = (_Float16)f1.w;
    }
    __syncthreads();
    v8h o;
#pragma unroll
    for (int i = 0; i < 8; ++i) o[i] = tile[c0 + i][r];
    *reinterpret_cast<v8h*>(Vth + ((size_t)bh * kN + d0 + r) * kT + t0 + c0) = o;
}

// ============================ TDM: contiguous 32KB K tile -> LDS =============
__device__ __forceinline__ void tdm_load_k_tile(const _Float16* gsrc, unsigned lds_off) {
    const unsigned long long ga = (unsigned long long)(const void*)gsrc;
    const unsigned ga_lo = (unsigned)__builtin_amdgcn_readfirstlane((int)(ga & 0xFFFFFFFFu));
    const unsigned ga_hi = (unsigned)__builtin_amdgcn_readfirstlane((int)(ga >> 32));
    const unsigned lds   = (unsigned)__builtin_amdgcn_readfirstlane((int)lds_off);
    uint4v g0;
    g0[0] = 1u;                                   // count=1, user descriptor
    g0[1] = lds;                                  // lds_addr
    g0[2] = ga_lo;                                // global_addr[31:0]
    g0[3] = (ga_hi & 0x01FFFFFFu) | (2u << 30);   // global_addr[56:32] | type=2
    int8v g1;
    g1[0] = (int)(1u << 16);                      // data_size = 1 -> 2 bytes/elem
    g1[1] = (int)0xFFFF0000u;                     // tensor_dim0 lo16 (huge)
    g1[2] = (int)0xFFFF7FFFu;                     // dim0 hi16 | tensor_dim1 lo16
    g1[3] = (int)((16384u << 16) | 0x7FFFu);      // tile_dim0=16384 | dim1 hi16
    g1[4] = 1;                                    // tile_dim1=1, tile_dim2=0
    g1[5] = 16384;                                // tensor_dim0_stride lo32
    g1[6] = 0;
    g1[7] = 0;
    int4v g2 = {0, 0, 0, 0};
    int4v g3 = {0, 0, 0, 0};
    int8v g4 = {0, 0, 0, 0, 0, 0, 0, 0};          // 6-arg toolchain variant
    __builtin_amdgcn_tensor_load_to_lds(g0, g1, g2, g3, g4, 0);
}

// ============================ Pass 2: flash attention on f16 =================
__global__ __launch_bounds__(128) void fa_f16_wmma(const _Float16* __restrict__ Qh,
                                                   const _Float16* __restrict__ Kh,
                                                   const _Float16* __restrict__ Vth,
                                                   float* __restrict__ out) {
    __shared__ __attribute__((aligned(64))) _Float16 Kbuf[2][KTILE * kN];   // 2 x 32 KB
    __shared__ __attribute__((aligned(64))) float    Sred[NWAVE][32][16];   // 8 KB
    __shared__ __attribute__((aligned(64))) _Float16 Pl[QT * KTILE];        // 1 KB

    const int tid  = threadIdx.x;
    const int lane = tid & 31;
    const int w    = tid >> 5;
    const int mrow = lane & 15;
    const int ksel = lane >> 4;

    const int qt = blockIdx.x % (kT / QT);
    const int h  = (blockIdx.x / (kT / QT)) % kH;
    const int b  = blockIdx.x / ((kT / QT) * kH);
    const int q0 = qt * QT;
    const int D0 = w * DW;
    const size_t bh = (size_t)b * kH + h;

    const _Float16* Kbase = Kh  + bh * (size_t)kT * kN;   // [s][d]
    const _Float16* Vbase = Vth + bh * (size_t)kN * kT;   // [d][t]

    // ---- Q A-fragments (already RoPE'd f16) ----
    v16h qa[4];
    {
        const _Float16* qrow = Qh + (bh * kT + (size_t)(q0 + mrow)) * kN;
#pragma unroll
        for (int c = 0; c < 4; ++c) {
            const int d0 = D0 + c * 32 + ksel * 8;
            v8h lo = *reinterpret_cast<const v8h*>(qrow + d0);
            v8h hi = *reinterpret_cast<const v8h*>(qrow + d0 + 16);
#pragma unroll
            for (int i = 0; i < 8; ++i) { qa[c][i] = lo[i]; qa[c][8 + i] = hi[i]; }
        }
    }

    v8f o[8];
#pragma unroll
    for (int j = 0; j < 8; ++j) o[j] = zero8();
    float mrun[8], lrun[8];
#pragma unroll
    for (int r = 0; r < 8; ++r) { mrun[r] = -1e30f; lrun[r] = 0.0f; }

    const unsigned lds0 = (unsigned)(size_t)(&Kbuf[0][0]);
    const unsigned lds1 = (unsigned)(size_t)(&Kbuf[1][0]);

    const int nkt = (q0 + QT + KTILE - 1) / KTILE;

    if (w == 0) tdm_load_k_tile(Kbase, lds0);              // prologue: tile 0

    for (int kt = 0; kt < nkt; ++kt) {
        const int s0 = kt * KTILE;
        if (w == 0) {
            if (kt + 1 < nkt) {
                tdm_load_k_tile(Kbase + (size_t)(s0 + KTILE) * kN,
                                ((kt + 1) & 1) ? lds1 : lds0);
                __builtin_amdgcn_s_wait_tensorcnt(1);      // tile kt complete
            } else {
                __builtin_amdgcn_s_wait_tensorcnt(0);
            }
        }
        __syncthreads();                                   // K tile kt visible
        const _Float16* Kt = &Kbuf[kt & 1][0];

        // ---- partial S = Q K^T over this wave's 128 dims ----
        v8f acc[2] = {zero8(), zero8()};
#pragma unroll
        for (int n = 0; n < 2; ++n) {
            const int key = n * 16 + mrow;
#pragma unroll
            for (int c = 0; c < 4; ++c) {
                v8h klo = *reinterpret_cast<const v8h*>(Kt + key * kN + D0 + c * 32 + ksel * 8);
                v8h khi = *reinterpret_cast<const v8h*>(Kt + key * kN + D0 + c * 32 + ksel * 8 + 16);
                v16h kb;
#pragma unroll
                for (int i = 0; i < 8; ++i) { kb[i] = klo[i]; kb[8 + i] = khi[i]; }
                acc[n] = __builtin_amdgcn_wmma_f32_16x16x32_f16(
                    false, qa[c], false, kb, (short)0, acc[n], false, false);
            }
        }

        // ---- cross-wave reduction of S ----
        *reinterpret_cast<v8f*>(&Sred[w][lane][0]) = acc[0];
        *reinterpret_cast<v8f*>(&Sred[w][lane][8]) = acc[1];
        __syncthreads();
        v8f sA = zero8(), sB = zero8();
#pragma unroll
        for (int ww = 0; ww < NWAVE; ++ww) {
            sA += *reinterpret_cast<const v8f*>(&Sred[ww][lane][0]);
            sB += *reinterpret_cast<const v8f*>(&Sred[ww][lane][8]);
        }

        // ---- scale + causal mask + online softmax ----
        const int cg0 = s0 + mrow, cg1 = cg0 + 16;
        float p0[8], p1[8];
#pragma unroll
        for (int r = 0; r < 8; ++r) {
            const int qg = q0 + ksel * 8 + r;
            float a0 = sA[r] * kScale;
            float a1 = sB[r] * kScale;
            if (cg0 > qg) a0 = -1e30f;
            if (cg1 > qg) a1 = -1e30f;
            float vm = fmaxf(a0, a1);
#pragma unroll
            for (int off = 8; off >= 1; off >>= 1)
                vm = fmaxf(vm, __shfl_xor(vm, off, 32));
            const float mnew = fmaxf(mrun[r], vm);
            const float corr = __expf(mrun[r] - mnew);
            const float e0 = __expf(a0 - mnew);
            const float e1 = __expf(a1 - mnew);
            float ps = e0 + e1;
#pragma unroll
            for (int off = 8; off >= 1; off >>= 1)
                ps += __shfl_xor(ps, off, 32);
            lrun[r] = lrun[r] * corr + ps;
            mrun[r] = mnew;
            p0[r] = e0;
            p1[r] = e1;
#pragma unroll
            for (int j = 0; j < 8; ++j) o[j][r] *= corr;
        }

        if (w == 0) {                     // publish P (f16)
#pragma unroll
            for (int r = 0; r < 8; ++r) {
                const int M = ksel * 8 + r;
                Pl[M * KTILE + mrow]      = (_Float16)p0[r];
                Pl[M * KTILE + 16 + mrow] = (_Float16)p1[r];
            }
        }
        __syncthreads();

        v16h pa;
        {
            v8h lo = *reinterpret_cast<const v8h*>(&Pl[mrow * KTILE + ksel * 16]);
            v8h hi = *reinterpret_cast<const v8h*>(&Pl[mrow * KTILE + ksel * 16 + 8]);
#pragma unroll
            for (int i = 0; i < 8; ++i) { pa[i] = lo[i]; pa[8 + i] = hi[i]; }
        }

        // ---- O += P * V  (V B-frags: 32 contiguous bytes from transposed f16) --
#pragma unroll
        for (int j = 0; j < 8; ++j) {
            const _Float16* vp = Vbase + (size_t)(D0 + j * 16 + mrow) * kT + s0 + ksel * 16;
            if (kt + 1 < nkt) __builtin_prefetch(vp + KTILE, 0, 1);
            v16h vb = *reinterpret_cast<const v16h*>(vp);
            o[j] = __builtin_amdgcn_wmma_f32_16x16x32_f16(
                false, pa, false, vb, (short)0, o[j], false, false);
        }
        __syncthreads();                  // protect Sred/Pl/Kbuf reuse
    }

    // ---- epilogue ----
#pragma unroll
    for (int r = 0; r < 8; ++r) {
        const float inv = 1.0f / lrun[r];
        const int qg = q0 + ksel * 8 + r;
        float* orow = out + ((size_t)b * kT + qg) * (kH * kN) + h * kN;
#pragma unroll
        for (int j = 0; j < 8; ++j)
            orow[D0 + j * 16 + mrow] = o[j][r] * inv;
    }
}

// ============================ Fallback: fused single pass (round-1) ==========
__global__ __launch_bounds__(128) void fa_rope_wmma(const float* __restrict__ Q,
                                                    const float* __restrict__ K,
                                                    const float* __restrict__ V,
                                                    float* __restrict__ out) {
    __shared__ __attribute__((aligned(64))) _Float16 Vt[kN * KTILE];
    __shared__ __attribute__((aligned(64))) float    Sred[NWAVE][32][16];
    __shared__ __attribute__((aligned(64))) _Float16 Pl[QT * KTILE];

    const int tid  = threadIdx.x;
    const int lane = tid & 31;
    const int w    = tid >> 5;
    const int mrow = lane & 15;
    const int ksel = lane >> 4;
    const int qt = blockIdx.x % (kT / QT);
    const int h  = (blockIdx.x / (kT / QT)) % kH;
    const int b  = blockIdx.x / ((kT / QT) * kH);
    const int q0 = qt * QT;
    const int D0 = w * DW;
    const size_t rowstride = (size_t)kH * kN;
    const size_t basebh    = (size_t)b * kT * rowstride + (size_t)h * kN;

    v16h qa[4];
    {
        const float* qrow = Q + basebh + (size_t)(q0 + mrow) * rowstride;
        const float  tq   = (float)(q0 + mrow);
#pragma unroll
        for (int c = 0; c < 4; ++c) {
#pragma unroll
            for (int half = 0; half < 2; ++half) {
                int d0 = D0 + c * 32 + ksel * 16 + half * 8;
                float4 f0 = *reinterpret_cast<const float4*>(qrow + d0);
                float4 f1 = *reinterpret_cast<const float4*>(qrow + d0 + 4);
                float vin[8] = {f0.x, f0.y, f0.z, f0.w, f1.x, f1.y, f1.z, f1.w};
#pragma unroll
                for (int p = 0; p < 4; ++p) {
                    float oe, oo;
                    rope_pair(vin[2 * p], vin[2 * p + 1], tq, pair_freq(d0 + 2 * p), oe, oo);
                    qa[c][half * 8 + 2 * p]     = (_Float16)oe;
                    qa[c][half * 8 + 2 * p + 1] = (_Float16)oo;
                }
            }
        }
    }
    float kfreq[4][8];
#pragma unroll
    for (int c = 0; c < 4; ++c)
#pragma unroll
        for (int p = 0; p < 8; ++p)
            kfreq[c][p] = pair_freq(D0 + c * 32 + ksel * 16 + 2 * p);

    v8f o[8];
#pragma unroll
    for (int j = 0; j < 8; ++j) o[j] = zero8();
    float mrun[8], lrun[8];
#pragma unroll
    for (int r = 0; r < 8; ++r) { mrun[r] = -1e30f; lrun[r] = 0.0f; }

    const int nkt = (q0 + QT + KTILE - 1) / KTILE;
    for (int kt = 0; kt < nkt; ++kt) {
        const int s0 = kt * KTILE;
        {
            const int srow  = tid >> 2;
            const int cbase = (tid & 3) * 128;
            const float* vrow = V + basebh + (size_t)(s0 + srow) * rowstride;
#pragma unroll 4
            for (int cc = 0; cc < 128; cc += 4) {
                int d = cbase + cc;
                float4 f = *reinterpret_cast<const float4*>(vrow + d);
                Vt[(d + 0) * KTILE + srow] = (_Float16)f.x;
                Vt[(d + 1) * KTILE + srow] = (_Float16)f.y;
                Vt[(d + 2) * KTILE + srow] = (_Float16)f.z;
                Vt[(d + 3) * KTILE + srow] = (_Float16)f.w;
            }
        }
        __syncthreads();

        v8f acc[2] = {zero8(), zero8()};
#pragma unroll
        for (int n = 0; n < 2; ++n) {
            const int   key = s0 + n * 16 + mrow;
            const float tk  = (float)key;
            const float* krow = K + basebh + (size_t)key * rowstride;
#pragma unroll
            for (int c = 0; c < 4; ++c) {
                const float* kp = krow + D0 + c * 32 + ksel * 16;
                float4 f0 = *reinterpret_cast<const float4*>(kp);
                float4 f1 = *reinterpret_cast<const float4*>(kp + 4);
                float4 f2 = *reinterpret_cast<const float4*>(kp + 8);
                float4 f3 = *reinterpret_cast<const float4*>(kp + 12);
                float vin[16] = {f0.x, f0.y, f0.z, f0.w, f1.x, f1.y, f1.z, f1.w,
                                 f2.x, f2.y, f2.z, f2.w, f3.x, f3.y, f3.z, f3.w};
                v16h kb;
#pragma unroll
                for (int p = 0; p < 8; ++p) {
                    float oe, oo;
                    rope_pair(vin[2 * p], vin[2 * p + 1], tk, kfreq[c][p], oe, oo);
                    kb[2 * p]     = (_Float16)oe;
                    kb[2 * p + 1] = (_Float16)oo;
                }
                acc[n] = __builtin_amdgcn_wmma_f32_16x16x32_f16(
                    false, qa[c], false, kb, (short)0, acc[n], false, false);
            }
        }
        *reinterpret_cast<v8f*>(&Sred[w][lane][0]) = acc[0];
        *reinterpret_cast<v8f*>(&Sred[w][lane][8]) = acc[1];
        __syncthreads();
        v8f sA = zero8(), sB = zero8();
#pragma unroll
        for (int ww = 0; ww < NWAVE; ++ww) {
            sA += *reinterpret_cast<const v8f*>(&Sred[ww][lane][0]);
            sB += *reinterpret_cast<const v8f*>(&Sred[ww][lane][8]);
        }
        const int cg0 = s0 + mrow, cg1 = cg0 + 16;
        float p0[8], p1[8];
#pragma unroll
        for (int r = 0; r < 8; ++r) {
            const int qg = q0 + ksel * 8 + r;
            float a0 = sA[r] * kScale;
            float a1 = sB[r] * kScale;
            if (cg0 > qg) a0 = -1e30f;
            if (cg1 > qg) a1 = -1e30f;
            float vm = fmaxf(a0, a1);
#pragma unroll
            for (int off = 8; off >= 1; off >>= 1)
                vm = fmaxf(vm, __shfl_xor(vm, off, 32));
            const float mnew = fmaxf(mrun[r], vm);
            const float corr = __expf(mrun[r] - mnew);
            const float e0 = __expf(a0 - mnew);
            const float e1 = __expf(a1 - mnew);
            float ps = e0 + e1;
#pragma unroll
            for (int off = 8; off >= 1; off >>= 1)
                ps += __shfl_xor(ps, off, 32);
            lrun[r] = lrun[r] * corr + ps;
            mrun[r] = mnew;
            p0[r] = e0; p1[r] = e1;
#pragma unroll
            for (int j = 0; j < 8; ++j) o[j][r] *= corr;
        }
        if (w == 0) {
#pragma unroll
            for (int r = 0; r < 8; ++r) {
                const int M = ksel * 8 + r;
                Pl[M * KTILE + mrow]      = (_Float16)p0[r];
                Pl[M * KTILE + 16 + mrow] = (_Float16)p1[r];
            }
        }
        __syncthreads();
        v16h pa;
        {
            v8h lo = *reinterpret_cast<const v8h*>(&Pl[mrow * KTILE + ksel * 16]);
            v8h hi = *reinterpret_cast<const v8h*>(&Pl[mrow * KTILE + ksel * 16 + 8]);
#pragma unroll
            for (int i = 0; i < 8; ++i) { pa[i] = lo[i]; pa[8 + i] = hi[i]; }
        }
#pragma unroll
        for (int j = 0; j < 8; ++j) {
            const int d = D0 + j * 16 + mrow;
            v16h vb = *reinterpret_cast<const v16h*>(&Vt[d * KTILE + ksel * 16]);
            o[j] = __builtin_amdgcn_wmma_f32_16x16x32_f16(
                false, pa, false, vb, (short)0, o[j], false, false);
        }
        __syncthreads();
    }
#pragma unroll
    for (int r = 0; r < 8; ++r) {
        const float inv = 1.0f / lrun[r];
        const int qg = q0 + ksel * 8 + r;
        float* orow = out + basebh + (size_t)qg * rowstride;
#pragma unroll
        for (int j = 0; j < 8; ++j)
            orow[D0 + j * 16 + mrow] = o[j][r] * inv;
    }
}

// ============================ host launcher ==================================
extern "C" void kernel_launch(void* const* d_in, const int* in_sizes, int n_in,
                              void* d_out, int out_size, void* d_ws, size_t ws_size,
                              hipStream_t stream) {
    (void)in_sizes; (void)n_in; (void)out_size;
    const float* Q = (const float*)d_in[0];
    const float* K = (const float*)d_in[1];
    const float* V = (const float*)d_in[2];
    float* out = (float*)d_out;

    const size_t elems = (size_t)kB * kH * kT * kN;          // 33.5M
    const size_t need  = elems * sizeof(_Float16) * 3;       // ~201 MB

    dim3 fgrid(kB * kH * (kT / QT));   // 4096 workgroups
    dim3 fblk(128);

    if (ws_size >= need) {
        _Float16* Qh  = (_Float16*)d_ws;
        _Float16* Kh  = Qh + elems;
        _Float16* Vth = Kh + elems;
        hipLaunchKernelGGL(rope_qk_f16, dim3(kB * kT), dim3(256), 0, stream, Q, K, Qh, Kh);
        hipLaunchKernelGGL(v_tr_f16, dim3(kB * kH * (kT / 32) * (kN / 32)), dim3(128),
                           0, stream, V, Vth);
        hipLaunchKernelGGL(fa_f16_wmma, fgrid, fblk, 0, stream, Qh, Kh, Vth, out);
    } else {
        hipLaunchKernelGGL(fa_rope_wmma, fgrid, fblk, 0, stream, Q, K, V, out);
    }
}
